// SimpleRNN_19043884990998
// MI455X (gfx1250) — matrix operand order repaired and checked by reading, not compiled, not run on hardware
//
#include <hip/hip_runtime.h>
#include <math.h>

// SimpleRNN on MI455X (gfx1250):
//  - 2047 sequential GEMV steps; Wh is repacked ONCE to bf16 in a contiguous-k (half h: k = 16h + s)
//    32x16 WMMA B-operand lane layout (32MB, L2-resident in the 192MB L2).
//  - Per step each wave streams its tile with global_load_b128 pairs straight
//    into v_wmma_f32_16x16x32_bf16 (fp32 accumulate), h staged in LDS as
//    packed bf16 pairs (ds_load_b128 for the A operand, row M=0 only).
//  - One persistent kernel + sense-reversal grid barrier (no launch overhead).
//  - Output projection uses the fp32 v_wmma_f32_16x16x4_f32 path (one-shot).

#define N_IN    1024
#define N_HIDN  4096
#define N_OUT   1024
#define SEQ     2048
#define NSTEPS  (SEQ - 1)      // 2047 recurrent steps

#define NBLOCKS  128
#define NTHREADS 256           // 8 waves of 32

#define NTILES     (N_HIDN / 16)          // 256 column tiles
#define NCHUNKS    (N_HIDN / 32)          // 128 K-chunks of 32
#define PACK_DW    ((size_t)NTILES * NCHUNKS * 256)   // 8,388,608 dwords = 32MB

typedef __attribute__((ext_vector_type(2)))  float    v2f;
typedef __attribute__((ext_vector_type(8)))  float    v8f;
typedef __attribute__((ext_vector_type(4)))  float    f32x4;
typedef __attribute__((ext_vector_type(4)))  unsigned u32x4;
typedef __attribute__((ext_vector_type(16))) __bf16   v16bf;

union V16 { u32x4 q[2]; v16bf v; };

// Round-to-nearest-even pack of two fp32 into packed bf16 pair (lo, hi).
__device__ __forceinline__ unsigned pack_bf16(float lo, float hi) {
  unsigned a = __float_as_uint(lo), b = __float_as_uint(hi);
  a = a + 0x7FFFu + ((a >> 16) & 1u);
  b = b + 0x7FFFu + ((b >> 16) & 1u);
  return (a >> 16) | (b & 0xFFFF0000u);
}

// ---------------- fp32 16x16x4 helper (output projection + fallback) -------
__device__ __forceinline__ float wmma_dot16(const float* __restrict__ B,
                                            size_t ldb,
                                            const float* hvec,
                                            int kbeg, int kend,
                                            int n0, int lane) {
  v8f acc = {};
  const int  col  = n0 + (lane & 15);
  const int  hi2  = (lane >> 4) << 1;      // 0 (lanes 0-15) or 2 (lanes 16-31)
  const bool lead = ((lane & 15) == 0);    // lanes 0 and 16 carry A-row M=0
  const float* bp = B + (size_t)(kbeg + hi2) * ldb + (size_t)col;
#pragma unroll 4
  for (int k = kbeg; k < kend; k += 4) {
    v2f a, b;
    b.x = bp[0];
    b.y = bp[ldb];
    float h0 = hvec[k + hi2];
    float h1 = hvec[k + 1 + hi2];
    a.x = lead ? h0 : 0.0f;
    a.y = lead ? h1 : 0.0f;
    acc = __builtin_amdgcn_wmma_f32_16x16x4_f32(false, a, false, b,
                                                (short)0, acc, false, false);
    bp += 4 * ldb;
  }
  return acc[0];   // D[M=0][N=lane] for lanes 0..15
}

// ---------------- grid barrier --------------------------------------------
__device__ __forceinline__ void grid_barrier(unsigned* cnt, unsigned* sns,
                                             unsigned* my_sense) {
  __builtin_amdgcn_fence(__ATOMIC_RELEASE, "agent");
  __syncthreads();
  if (threadIdx.x == 0) {
    unsigned next = *my_sense + 1u;
    unsigned prev = __hip_atomic_fetch_add(cnt, 1u, __ATOMIC_ACQ_REL,
                                           __HIP_MEMORY_SCOPE_AGENT);
    if (prev == (unsigned)(NBLOCKS - 1)) {
      __hip_atomic_store(cnt, 0u, __ATOMIC_RELAXED, __HIP_MEMORY_SCOPE_AGENT);
      __hip_atomic_store(sns, next, __ATOMIC_RELEASE, __HIP_MEMORY_SCOPE_AGENT);
    } else {
      while (__hip_atomic_load(sns, __ATOMIC_ACQUIRE,
                               __HIP_MEMORY_SCOPE_AGENT) != next) {
        __builtin_amdgcn_s_sleep(8);
      }
    }
    *my_sense = next;
  }
  __syncthreads();
  __builtin_amdgcn_fence(__ATOMIC_ACQUIRE, "agent");
}

__global__ void rnn_init(float* h0, float* h1, unsigned* bar) {
  int i = blockIdx.x * blockDim.x + threadIdx.x;
  if (i < N_HIDN) { h0[i] = 0.0f; h1[i] = 0.0f; }
  if (i == 0) { bar[0] = 0u; bar[1] = 0u; }
}

// ---------------- one-time Wh fp32 -> bf16 pack into WMMA B-operand layout -
// Block b = tile*NCHUNKS + chunk.  Output block: 256 dwords,
// dword index = lane*8 + v;  lane<16: rows K=2v,2v+1, col=lane;
//                            lane>=16: rows K=16+2v,17+2v, col=lane-16.
__global__ __launch_bounds__(64)
void pack_wh(const float* __restrict__ w_hid, unsigned* __restrict__ out) {
  __shared__ float tile[32 * 16];
  const float* wh = w_hid + (size_t)N_IN * N_HIDN;
  const int blk = blockIdx.x;
  const int t   = blk >> 7;          // tile index (16 columns)
  const int c   = blk & (NCHUNKS - 1);
  const int tid = threadIdx.x;
  for (int e = tid; e < 512; e += 64) {
    int r = e >> 4, col = e & 15;
    tile[e] = wh[(size_t)(c * 32 + r) * N_HIDN + t * 16 + col];
  }
  __syncthreads();
  unsigned* ob = out + (size_t)blk * 256;
  for (int i = tid; i < 256; i += 64) {
    int v    = i & 7;
    int lane = i >> 3;
    int r    = ((lane >> 4) << 4) + 2 * v;
    int col  = lane & 15;
    ob[i] = pack_bf16(tile[r * 16 + col], tile[(r + 1) * 16 + col]);
  }
}

// ---------------- bf16 persistent kernel ----------------------------------
__global__ __launch_bounds__(NTHREADS)
void rnn_persistent_bf16(const int* __restrict__ xs,
                         const float* __restrict__ w_hid,
                         const float* __restrict__ b_hid,
                         const float* __restrict__ w_out,
                         const float* __restrict__ b_out,
                         const unsigned* __restrict__ whp,  // packed bf16 Wh
                         float* __restrict__ out,
                         float* __restrict__ h_a,
                         float* __restrict__ h_b,
                         float* __restrict__ logits,
                         unsigned* __restrict__ bar_cnt,
                         unsigned* __restrict__ bar_sns) {
  __shared__ __align__(16) unsigned sh_hbf[N_HIDN / 2 + 16]; // packed h + zeros
  __shared__ float sh_h[N_HIDN];       // fp32 h for the output projection
  __shared__ float sh_part[8][16];
  __shared__ float sh_red[NTHREADS];

  const int tid  = threadIdx.x;
  const int wave = tid >> 5;
  const int lane = tid & 31;
  const bool lead = ((lane & 15) == 0);
  unsigned my_sense = 0u;

  if (tid < 16) sh_hbf[N_HIDN / 2 + tid] = 0u;   // zero region for A operand

  const float* wx = w_hid;                              // rows [0,1024)
  float* hc = h_a;   // zero-initialized by rnn_init
  float* hn = h_b;

  const int tile = blockIdx.x * 2 + (wave >> 2);  // 2 tiles/block, 4 waves each
  const int kq   = wave & 3;                      // K quarter (1024 k)

  for (int t = 0; t < NSTEPS; ++t) {
    // Stage h into LDS as packed bf16 pairs (pair j = h[2j],h[2j+1]).
    for (int i = tid; i < N_HIDN / 4; i += NTHREADS) {
      f32x4 hv = ((const f32x4*)hc)[i];
      sh_hbf[2 * i]     = pack_bf16(hv.x, hv.y);
      sh_hbf[2 * i + 1] = pack_bf16(hv.z, hv.w);
    }
    __syncthreads();

    // B stream: 1KB/chunk, 2 x b128 per lane.  A: 2 x b128 from LDS (row 0).
    const u32x4* bptr  = (const u32x4*)whp +
                         ((size_t)tile * NCHUNKS + kq * 32) * 64 + lane * 2;
    const u32x4* abase = (const u32x4*)sh_hbf + kq * 128 + ((lane >> 4) & 1) * 2;
    const u32x4* azero = (const u32x4*)(sh_hbf + N_HIDN / 2);
    v8f acc = {};
#pragma unroll 2
    for (int c = 0; c < 32; ++c) {
      V16 B, A;
      B.q[0] = bptr[0];
      B.q[1] = bptr[1];
      const u32x4* ap = lead ? abase : azero;
      A.q[0] = ap[0];
      A.q[1] = ap[1];
      acc = __builtin_amdgcn_wmma_f32_16x16x32_bf16(false, A.v, false, B.v,
                                                    (short)0, acc, false, false);
      bptr  += 64;
      abase += 4;
    }
    float p = acc[0];
    if (lane < 16) sh_part[wave][lane] = p;
    __syncthreads();

    if (wave == 0) {
      const int local = lane & 15;
      const int which = lane >> 4;
      float s = sh_part[which * 4 + 0][local] + sh_part[which * 4 + 1][local] +
                sh_part[which * 4 + 2][local] + sh_part[which * 4 + 3][local];
      const int colg = (blockIdx.x * 2 + which) * 16 + local;
      const int x    = xs[t];
      s += wx[(size_t)x * N_HIDN + colg] + b_hid[colg];
      hn[colg] = s;
    }

    grid_barrier(bar_cnt, bar_sns, &my_sense);
    float* tmp = hc; hc = hn; hn = tmp;
  }

  // ---- Output projection (one-shot, fp32 WMMA path) ----
  if (blockIdx.x < N_OUT / 16) {
    for (int i = tid; i < N_HIDN / 4; i += NTHREADS)
      ((f32x4*)sh_h)[i] = ((const f32x4*)hc)[i];
    __syncthreads();

    const int n0 = blockIdx.x * 16;
    const float* wo = w_out + (size_t)N_IN * N_OUT;
    float p = wmma_dot16(wo, (size_t)N_OUT, sh_h,
                         wave * (N_HIDN / 8), (wave + 1) * (N_HIDN / 8),
                         n0, lane);
    if (lane < 16) sh_part[wave][lane] = p;
    __syncthreads();

    if (wave == 0 && lane < 16) {
      float s = 0.0f;
#pragma unroll
      for (int w2 = 0; w2 < 8; ++w2) s += sh_part[w2][lane];
      const int colg = n0 + lane;
      const int x    = xs[SEQ - 1];
      s += w_out[(size_t)x * N_OUT + colg] + b_out[colg];
      logits[colg] = s;
    }
  }

  grid_barrier(bar_cnt, bar_sns, &my_sense);

  if (blockIdx.x == 0) {
    float vmax = -INFINITY;
    for (int i = tid; i < N_OUT; i += NTHREADS) vmax = fmaxf(vmax, logits[i]);
    sh_red[tid] = vmax;
    __syncthreads();
    for (int s = NTHREADS / 2; s > 0; s >>= 1) {
      if (tid < s) sh_red[tid] = fmaxf(sh_red[tid], sh_red[tid + s]);
      __syncthreads();
    }
    vmax = sh_red[0];
    __syncthreads();
    float vsum = 0.0f;
    for (int i = tid; i < N_OUT; i += NTHREADS) vsum += __expf(logits[i] - vmax);
    sh_red[tid] = vsum;
    __syncthreads();
    for (int s = NTHREADS / 2; s > 0; s >>= 1) {
      if (tid < s) sh_red[tid] += sh_red[tid + s];
      __syncthreads();
    }
    const float lse = vmax + __logf(sh_red[0]);
    for (int i = tid; i < N_OUT; i += NTHREADS) out[i] = logits[i] - lse;
  }
}

// ---------------- fp32 fallback persistent kernel (ws too small) ----------
__global__ __launch_bounds__(NTHREADS)
void rnn_persistent_f32(const int* __restrict__ xs,
                        const float* __restrict__ w_hid,
                        const float* __restrict__ b_hid,
                        const float* __restrict__ w_out,
                        const float* __restrict__ b_out,
                        float* __restrict__ out,
                        float* __restrict__ h_a,
                        float* __restrict__ h_b,
                        float* __restrict__ logits,
                        unsigned* __restrict__ bar_cnt,
                        unsigned* __restrict__ bar_sns) {
  __shared__ float sh_h[N_HIDN];
  __shared__ float sh_part[8][16];
  __shared__ float sh_red[NTHREADS];

  const int tid  = threadIdx.x;
  const int wave = tid >> 5;
  const int lane = tid & 31;
  unsigned my_sense = 0u;

  const float* wx = w_hid;
  const float* wh = w_hid + (size_t)N_IN * N_HIDN;
  float* hc = h_a;
  float* hn = h_b;

  for (int t = 0; t < NSTEPS; ++t) {
    for (int i = tid; i < N_HIDN / 4; i += NTHREADS)
      ((f32x4*)sh_h)[i] = ((const f32x4*)hc)[i];
    __syncthreads();

    const int tile = blockIdx.x * 2 + (wave >> 2);
    const int kq   = wave & 3;
    float p = wmma_dot16(wh, (size_t)N_HIDN, sh_h,
                         kq * (N_HIDN / 4), (kq + 1) * (N_HIDN / 4),
                         tile * 16, lane);
    if (lane < 16) sh_part[wave][lane] = p;
    __syncthreads();

    if (wave == 0) {
      const int local = lane & 15;
      const int which = lane >> 4;
      float s = sh_part[which * 4 + 0][local] + sh_part[which * 4 + 1][local] +
                sh_part[which * 4 + 2][local] + sh_part[which * 4 + 3][local];
      const int colg = (blockIdx.x * 2 + which) * 16 + local;
      const int x    = xs[t];
      s += wx[(size_t)x * N_HIDN + colg] + b_hid[colg];
      hn[colg] = s;
    }

    grid_barrier(bar_cnt, bar_sns, &my_sense);
    float* tmp = hc; hc = hn; hn = tmp;
  }

  if (blockIdx.x < N_OUT / 16) {
    for (int i = tid; i < N_HIDN / 4; i += NTHREADS)
      ((f32x4*)sh_h)[i] = ((const f32x4*)hc)[i];
    __syncthreads();
    const int n0 = blockIdx.x * 16;
    const float* wo = w_out + (size_t)N_IN * N_OUT;
    float p = wmma_dot16(wo, (size_t)N_OUT, sh_h,
                         wave * (N_HIDN / 8), (wave + 1) * (N_HIDN / 8),
                         n0, lane);
    if (lane < 16) sh_part[wave][lane] = p;
    __syncthreads();
    if (wave == 0 && lane < 16) {
      float s = 0.0f;
#pragma unroll
      for (int w2 = 0; w2 < 8; ++w2) s += sh_part[w2][lane];
      const int colg = n0 + lane;
      const int x    = xs[SEQ - 1];
      s += w_out[(size_t)x * N_OUT + colg] + b_out[colg];
      logits[colg] = s;
    }
  }

  grid_barrier(bar_cnt, bar_sns, &my_sense);

  if (blockIdx.x == 0) {
    float vmax = -INFINITY;
    for (int i = tid; i < N_OUT; i += NTHREADS) vmax = fmaxf(vmax, logits[i]);
    sh_red[tid] = vmax;
    __syncthreads();
    for (int s = NTHREADS / 2; s > 0; s >>= 1) {
      if (tid < s) sh_red[tid] = fmaxf(sh_red[tid], sh_red[tid + s]);
      __syncthreads();
    }
    vmax = sh_red[0];
    __syncthreads();
    float vsum = 0.0f;
    for (int i = tid; i < N_OUT; i += NTHREADS) vsum += __expf(logits[i] - vmax);
    sh_red[tid] = vsum;
    __syncthreads();
    for (int s = NTHREADS / 2; s > 0; s >>= 1) {
      if (tid < s) sh_red[tid] += sh_red[tid + s];
      __syncthreads();
    }
    const float lse = vmax + __logf(sh_red[0]);
    for (int i = tid; i < N_OUT; i += NTHREADS) out[i] = logits[i] - lse;
  }
}

extern "C" void kernel_launch(void* const* d_in, const int* in_sizes, int n_in,
                              void* d_out, int out_size, void* d_ws, size_t ws_size,
                              hipStream_t stream) {
  (void)in_sizes; (void)n_in; (void)out_size;
  const int*   xs    = (const int*)d_in[0];
  const float* w_hid = (const float*)d_in[1];
  const float* b_hid = (const float*)d_in[2];
  const float* w_out = (const float*)d_in[3];
  const float* b_out = (const float*)d_in[4];
  float*       out   = (float*)d_out;

  const size_t tail_dw  = (size_t)N_HIDN * 2 + N_OUT + 2;     // h0,h1,logits,bar
  const size_t need_bf  = (PACK_DW + tail_dw) * sizeof(unsigned);

  if (ws_size >= need_bf) {
    unsigned* whp    = (unsigned*)d_ws;              // 32MB packed bf16 Wh
    float*    h0     = (float*)(whp + PACK_DW);
    float*    h1     = h0 + N_HIDN;
    float*    logits = h1 + N_HIDN;
    unsigned* bar    = (unsigned*)(logits + N_OUT);

    pack_wh<<<NTILES * NCHUNKS, 64, 0, stream>>>(w_hid, whp);
    rnn_init<<<(N_HIDN + 255) / 256, 256, 0, stream>>>(h0, h1, bar);
    rnn_persistent_bf16<<<NBLOCKS, NTHREADS, 0, stream>>>(
        xs, w_hid, b_hid, w_out, b_out, whp, out, h0, h1, logits, bar, bar + 1);
  } else {
    float*    h0     = (float*)d_ws;
    float*    h1     = h0 + N_HIDN;
    float*    logits = h1 + N_HIDN;
    unsigned* bar    = (unsigned*)(logits + N_OUT);

    rnn_init<<<(N_HIDN + 255) / 256, 256, 0, stream>>>(h0, h1, bar);
    rnn_persistent_f32<<<NBLOCKS, NTHREADS, 0, stream>>>(
        xs, w_hid, b_hid, w_out, b_out, out, h0, h1, logits, bar, bar + 1);
  }
}
